// TransposableGene_62285615726973
// MI455X (gfx1250) — compile-verified, matrix operand order checked by reading, not run on hardware
//
#include <hip/hip_runtime.h>
#include <hip/hip_bf16.h>

#define N_NODES  50000
#define N_EDGES  1600000
#define N_GRAPHS 64
#define F_IN     256
#define HDIM     1024
#define EPS_LN   1e-5f

typedef __attribute__((ext_vector_type(16))) __bf16 v16bf;
typedef __attribute__((ext_vector_type(8)))  float  v8f;

__device__ __forceinline__ unsigned short f2bf(float f) {
    unsigned int u = __float_as_uint(f);
    u += 0x7FFFu + ((u >> 16) & 1u);   // round-to-nearest-even
    return (unsigned short)(u >> 16);
}
__device__ __forceinline__ __bf16 bfbits(unsigned short u) {
    union { unsigned short s; __bf16 b; } cv; cv.s = u; return cv.b;
}

// CDNA5 async global->LDS copy: 16 bytes per lane, tracked with ASYNCcnt.
__device__ __forceinline__ unsigned ldsOff(const void* p) {
    return (unsigned)(unsigned long long)p;
}
__device__ __forceinline__ void async_cp16(unsigned lds, const void* gsrc) {
    asm volatile("global_load_async_to_lds_b128 %0, %1, off"
                 :: "v"(lds), "v"(gsrc) : "memory");
}
__device__ __forceinline__ void wait_async3() {
    asm volatile("s_wait_asynccnt 0x3" ::: "memory");
}
__device__ __forceinline__ void wait_async0() {
    asm volatile("s_wait_asynccnt 0x0" ::: "memory");
}

// ---------------------------------------------------------------------------
// Small elementwise helpers
// ---------------------------------------------------------------------------
__global__ void gcn_fill_f32(float* __restrict__ p, float val, int n) {
    int i = blockIdx.x * 256 + threadIdx.x;
    if (i < n) p[i] = val;
}

__global__ void gcn_convert_bf16(const float* __restrict__ s, unsigned short* __restrict__ d, int n) {
    int i = blockIdx.x * 256 + threadIdx.x;
    if (i < n) d[i] = f2bf(s[i]);
}

// W is [K][H] row-major; write Wt[n*K + k] = bf16(W[k*H + n]).
__global__ void gcn_convert_transpose(const float* __restrict__ W, unsigned short* __restrict__ Wt, int K) {
    int i = blockIdx.x * 256 + threadIdx.x;
    if (i < K * HDIM) {
        int k = i / HDIM, n = i % HDIM;
        Wt[(size_t)n * K + k] = f2bf(W[i]);
    }
}

__global__ void gcn_deg_accum(const int* __restrict__ col, float* __restrict__ deg, int n) {
    int i = blockIdx.x * 256 + threadIdx.x;
    if (i < n) unsafeAtomicAdd(&deg[col[i]], 1.0f);
}

__global__ void gcn_to_dinv(float* __restrict__ d, int n) {
    int i = blockIdx.x * 256 + threadIdx.x;
    if (i < n) {
        float x = d[i];
        d[i] = (x > 0.0f) ? rsqrtf(x) : 0.0f;
    }
}

// ---------------------------------------------------------------------------
// bf16 WMMA GEMM:  C[M,HDIM] = Abf[M,K] @ Wt[HDIM,K]^T   (both bf16)
// Block tile 64x128, 8 waves (2x4), each wave 32x32 via four 16x16x32 WMMAs.
// Double-buffered LDS tiles staged with global_load_async_to_lds_b128; the
// next tile's DMA overlaps the current tile's WMMAs (ASYNCcnt pipeline).
// ---------------------------------------------------------------------------
#define BM 64
#define BN 128
#define KB 32
#define LDP 8

__global__ __launch_bounds__(256)
void gcn_gemm_bf16(const unsigned short* __restrict__ Abf,
                   const unsigned short* __restrict__ Wt,
                   float* __restrict__ C, int M, int K) {
    __shared__ __align__(16) unsigned short As[2][BM][KB + LDP];   // [m][k], row stride 80B
    __shared__ __align__(16) unsigned short Bs[2][BN][KB + LDP];   // [n][k]
    __shared__ __align__(16) unsigned short dummy[8];              // sink for tail waves

    const int tid  = threadIdx.x;
    const int lane = tid & 31;
    const int wave = tid >> 5;
    const int wm   = wave & 1;     // 0..1  (32-row slab)
    const int wn   = wave >> 1;    // 0..3  (32-col slab)
    const int lh   = lane >> 4;    // lane half
    const int l16  = lane & 15;
    const int mBase = blockIdx.x * BM;
    const int nBase = blockIdx.y * BN;

    // Per-thread tile-staging coordinates (16B chunks).
    const int aRow = tid >> 2, aCh = tid & 3;          // A: 64 rows x 4 chunks, 1/thread
    const int gmA  = mBase + aRow;
    const bool aOk = (gmA < M);

    // Issue the 3 async copies (1 A-chunk + 2 B-chunks) for tile kt into parity p.
    // Every wave issues exactly 3 so that s_wait_asynccnt immediates are uniform;
    // out-of-range A rows copy into a dummy LDS slot instead.
    auto issueTile = [&](int kt, int p) {
        const int kOff = kt * KB;
        unsigned aDst = aOk ? ldsOff(&As[p][aRow][aCh * 8]) : ldsOff(&dummy[0]);
        const unsigned short* aSrc = aOk ? (Abf + (size_t)gmA * K + kOff + aCh * 8) : Abf;
        async_cp16(aDst, aSrc);
        #pragma unroll
        for (int i = 0; i < 2; ++i) {
            int c   = tid * 2 + i;
            int row = c >> 2, ch = c & 3;
            async_cp16(ldsOff(&Bs[p][row][ch * 8]),
                       Wt + (size_t)(nBase + row) * K + kOff + ch * 8);
        }
    };

    // Prologue: zero the permanently-invalid A rows in both parities (never
    // overwritten afterwards), then kick off tile 0.
    if (!aOk) {
        uint4 z = {0u, 0u, 0u, 0u};
        *(uint4*)&As[0][aRow][aCh * 8] = z;
        *(uint4*)&As[1][aRow][aCh * 8] = z;
    }
    issueTile(0, 0);

    v8f acc[2][2];
    #pragma unroll
    for (int i = 0; i < 2; ++i)
        #pragma unroll
        for (int j = 0; j < 2; ++j)
            #pragma unroll
            for (int r = 0; r < 8; ++r) acc[i][j][r] = 0.0f;

    const int kTiles = K / KB;
    for (int kt = 0; kt < kTiles; ++kt) {
        const int p = kt & 1;
        if (kt + 1 < kTiles) {
            issueTile(kt + 1, p ^ 1);
            wait_async3();          // tile kt landed; tile kt+1 DMA overlaps compute
        } else {
            wait_async0();
        }
        __syncthreads();

        // Fragments per documented 16-bit layout:
        //   lane half lh: elements 0..7  -> K = lh*8 + e
        //                 elements 8..15 -> K = 16 + lh*8 + (e-8)
        v16bf afr[2], bfr[2];
        #pragma unroll
        for (int i = 0; i < 2; ++i) {
            int m = wm * 32 + i * 16 + l16;
            #pragma unroll
            for (int e = 0; e < 8; ++e) afr[i][e]     = bfbits(As[p][m][lh * 8 + e]);
            #pragma unroll
            for (int e = 0; e < 8; ++e) afr[i][8 + e] = bfbits(As[p][m][16 + lh * 8 + e]);
        }
        #pragma unroll
        for (int j = 0; j < 2; ++j) {
            int n = wn * 32 + j * 16 + l16;
            #pragma unroll
            for (int e = 0; e < 8; ++e) bfr[j][e]     = bfbits(Bs[p][n][lh * 8 + e]);
            #pragma unroll
            for (int e = 0; e < 8; ++e) bfr[j][8 + e] = bfbits(Bs[p][n][16 + lh * 8 + e]);
        }

        #pragma unroll
        for (int i = 0; i < 2; ++i)
            #pragma unroll
            for (int j = 0; j < 2; ++j)
                acc[i][j] = __builtin_amdgcn_wmma_f32_16x16x32_bf16(
                    false, afr[i], false, bfr[j], (short)0, acc[i][j], false, false);
        __syncthreads();   // WAR guard before parity p is refilled next iteration
    }

    // Store D: VGPR r, lane -> (m = r + 8*lh, n = lane&15) within each 16x16 tile.
    #pragma unroll
    for (int i = 0; i < 2; ++i) {
        #pragma unroll
        for (int j = 0; j < 2; ++j) {
            int n = nBase + wn * 32 + j * 16 + l16;
            #pragma unroll
            for (int r = 0; r < 8; ++r) {
                int m = mBase + wm * 32 + i * 16 + r + 8 * lh;
                if (m < M) C[(size_t)m * HDIM + n] = acc[i][j][r];
            }
        }
    }
}

// ---------------------------------------------------------------------------
// Edge scatter:  agg[col] += ht[row] * dinv[row]*dinv[col]   (self-loops appended)
// ---------------------------------------------------------------------------
__global__ __launch_bounds__(256)
void gcn_scatter(const float* __restrict__ ht, const float* __restrict__ dinv,
                 const int* __restrict__ ei, float* __restrict__ agg) {
    int e = blockIdx.x;
    int r, c;
    if (e < N_EDGES) { r = ei[e]; c = ei[N_EDGES + e]; }
    else             { r = c = e - N_EDGES; }
    float nrm = dinv[r] * dinv[c];
    const float* src = ht + (size_t)r * HDIM;
    float*       dst = agg + (size_t)c * HDIM;
    __builtin_prefetch(src, 0, 1);          // global_prefetch_b8
    #pragma unroll
    for (int j = 0; j < 4; ++j) {
        int f = threadIdx.x + j * 256;
        unsafeAtomicAdd(dst + f, src[f] * nrm);
    }
}

// ---------------------------------------------------------------------------
// Bias + LayerNorm + affine + ReLU (in place) + bf16 mirror for next GEMM.
// ---------------------------------------------------------------------------
__global__ __launch_bounds__(256)
void gcn_ln_relu(float* __restrict__ h, unsigned short* __restrict__ hbf,
                 const float* __restrict__ bias,
                 const float* __restrict__ gamma, const float* __restrict__ beta) {
    __shared__ float r1[256], r2[256];
    const int n = blockIdx.x, tid = threadIdx.x;
    float* row = h + (size_t)n * HDIM;
    unsigned short* rowbf = hbf + (size_t)n * HDIM;
    float v[4], s = 0.0f, sq = 0.0f;
    #pragma unroll
    for (int j = 0; j < 4; ++j) {
        int idx = tid + j * 256;
        v[j] = row[idx] + bias[idx];
        s  += v[j];
        sq += v[j] * v[j];
    }
    r1[tid] = s; r2[tid] = sq;
    __syncthreads();
    for (int st = 128; st > 0; st >>= 1) {
        if (tid < st) { r1[tid] += r1[tid + st]; r2[tid] += r2[tid + st]; }
        __syncthreads();
    }
    float mu   = r1[0] * (1.0f / HDIM);
    float var  = r2[0] * (1.0f / HDIM) - mu * mu;
    float rstd = rsqrtf(var + EPS_LN);
    #pragma unroll
    for (int j = 0; j < 4; ++j) {
        int idx = tid + j * 256;
        float hv = (v[j] - mu) * rstd * gamma[idx] + beta[idx];
        hv = fmaxf(hv, 0.0f);
        row[idx]   = hv;
        rowbf[idx] = f2bf(hv);
    }
}

// ---------------------------------------------------------------------------
// Graph pooling: atomic sum into out[64][1024] + counts, then divide.
// ---------------------------------------------------------------------------
__global__ __launch_bounds__(256)
void gcn_pool(const float* __restrict__ h, const int* __restrict__ batch,
              float* __restrict__ out, float* __restrict__ cnt) {
    int n = blockIdx.x;
    int g = batch[n];
    const float* row = h + (size_t)n * HDIM;
    #pragma unroll
    for (int j = 0; j < 4; ++j) {
        int f = threadIdx.x + j * 256;
        unsafeAtomicAdd(out + (size_t)g * HDIM + f, row[f]);
    }
    if (threadIdx.x == 0) unsafeAtomicAdd(cnt + g, 1.0f);
}

__global__ void gcn_div(float* __restrict__ out, const float* __restrict__ cnt) {
    int i = blockIdx.x * 256 + threadIdx.x;
    if (i < N_GRAPHS * HDIM) out[i] /= fmaxf(cnt[i >> 10], 1.0f);
}

// ---------------------------------------------------------------------------
// Launch
// ---------------------------------------------------------------------------
extern "C" void kernel_launch(void* const* d_in, const int* in_sizes, int n_in,
                              void* d_out, int out_size, void* d_ws, size_t ws_size,
                              hipStream_t stream) {
    const float* x      = (const float*)d_in[0];
    const int*   ei     = (const int*)d_in[1];
    const int*   batch  = (const int*)d_in[2];
    const float* W0     = (const float*)d_in[3];
    const float* b0     = (const float*)d_in[4];
    const float* Ws     = (const float*)d_in[5];
    const float* bs     = (const float*)d_in[6];
    const float* gammas = (const float*)d_in[7];
    const float* betas  = (const float*)d_in[8];
    float* out = (float*)d_out;

    char* ws = (char*)d_ws;
    size_t off = 0;
    auto alloc = [&](size_t bytes) { size_t o = off; off += (bytes + 255) & ~(size_t)255; return o; };

    float*          dinv = (float*)(ws + alloc((size_t)N_NODES * 4));
    float*          cnt  = (float*)(ws + alloc(256));
    unsigned short* wt   = (unsigned short*)(ws + alloc(((size_t)F_IN * HDIM + 3ull * HDIM * HDIM) * 2));
    unsigned short* xbf  = (unsigned short*)(ws + alloc((size_t)N_NODES * F_IN * 2));
    unsigned short* hbf  = (unsigned short*)(ws + alloc((size_t)N_NODES * HDIM * 2));
    float*          buf0 = (float*)(ws + alloc((size_t)N_NODES * HDIM * 4));
    float*          buf1 = (float*)(ws + alloc((size_t)N_NODES * HDIM * 4));

    const size_t nhBytes = (size_t)N_NODES * HDIM * 4;

    // 1. Weights -> bf16, transposed to [n][k]; x -> bf16.
    {
        int nW = F_IN * HDIM;
        gcn_convert_transpose<<<(nW + 255) / 256, 256, 0, stream>>>(W0, wt, F_IN);
        int nL = HDIM * HDIM;
        for (int i = 0; i < 3; ++i)
            gcn_convert_transpose<<<(nL + 255) / 256, 256, 0, stream>>>(
                Ws + (size_t)i * nL, wt + (size_t)F_IN * HDIM + (size_t)i * nL, HDIM);
        int nX = N_NODES * F_IN;
        gcn_convert_bf16<<<(nX + 255) / 256, 256, 0, stream>>>(x, xbf, nX);
    }

    // 2. Degrees (self-loop => init 1.0), then dinv = rsqrt(deg).
    gcn_fill_f32<<<(N_NODES + 255) / 256, 256, 0, stream>>>(dinv, 1.0f, N_NODES);
    gcn_deg_accum<<<(N_EDGES + 255) / 256, 256, 0, stream>>>(ei + N_EDGES, dinv, N_EDGES);
    gcn_to_dinv<<<(N_NODES + 255) / 256, 256, 0, stream>>>(dinv, N_NODES);

    const dim3 gemmGrid((N_NODES + BM - 1) / BM, HDIM / BN);
    const int  eTot = N_EDGES + N_NODES;

    // 3. Layer 0: xbf[N,256] @ Wt0 -> buf1; scatter -> buf0; LN+ReLU (+hbf).
    gcn_gemm_bf16<<<gemmGrid, 256, 0, stream>>>(xbf, wt, buf1, N_NODES, F_IN);
    hipMemsetAsync(buf0, 0, nhBytes, stream);
    gcn_scatter<<<eTot, 256, 0, stream>>>(buf1, dinv, ei, buf0);
    gcn_ln_relu<<<N_NODES, 256, 0, stream>>>(buf0, hbf, b0, gammas, betas);

    // 4. Layers 1..3.
    for (int i = 0; i < 3; ++i) {
        const unsigned short* wptr = wt + (size_t)F_IN * HDIM + (size_t)i * HDIM * HDIM;
        gcn_gemm_bf16<<<gemmGrid, 256, 0, stream>>>(hbf, wptr, buf1, N_NODES, HDIM);
        hipMemsetAsync(buf0, 0, nhBytes, stream);
        gcn_scatter<<<eTot, 256, 0, stream>>>(buf1, dinv, ei, buf0);
        gcn_ln_relu<<<N_NODES, 256, 0, stream>>>(buf0, hbf, bs + (size_t)i * HDIM,
                                                 gammas + (size_t)(i + 1) * HDIM,
                                                 betas + (size_t)(i + 1) * HDIM);
    }

    // 5. Graph pooling (scatter-mean over batch).
    hipMemsetAsync(out, 0, (size_t)N_GRAPHS * HDIM * 4, stream);
    hipMemsetAsync(cnt, 0, 256, stream);
    gcn_pool<<<N_NODES, 256, 0, stream>>>(buf0, batch, out, cnt);
    gcn_div<<<(N_GRAPHS * HDIM + 255) / 256, 256, 0, stream>>>(out, cnt);
}
